// TransducerBeamSearcher_58265526337843
// MI455X (gfx1250) — matrix operand order repaired and checked by reading, not compiled, not run on hardware
//
#include <hip/hip_runtime.h>
#include <cstdint>
#include <cstddef>

// ---------------------------------------------------------------------------
// Transducer greedy decode on MI455X (gfx1250, wave32, WMMA).
//
// Strategy:
//  * One-time: repack Wc (640x5000), [Wx;Wh] (1280x2560) and embed to bf16 in
//    native V_WMMA_F32_16X16X32_BF16 fragment layout (~19 MB, L2-resident).
//  * Per step t (1000 serial steps, all launched on `stream` so the harness
//    graph-captures the full loop):
//      GEMM1  : joint[32x640] @ WcPack  -> logits[32x5008]   (WMMA bf16)
//      REDUCE : online log-softmax/argmax per row, token select, score accum,
//               pack [embed[tok]; h] into A-fragments for the LSTM GEMM
//      GEMM2  : xcat[32x1280] @ W2Pack -> gates[32x2560]     (WMMA bf16)
//      GATES  : sigmoid/tanh cell update (masked), pack tanh(tn_{t+1}+out_PN)
//               A-fragments for the next step
//  * Workspace requirement: ~21 MB.
// ---------------------------------------------------------------------------

typedef __attribute__((ext_vector_type(16))) __bf16 v16bf;
typedef __attribute__((ext_vector_type(8)))  float  v8f;

#define BATCH 32
#define TLEN  1000
#define HDIM  640
#define VDIM  5000
#define H4    2560   // 4*H
#define K2DIM 1280   // 2*H
#define VPAD  5008   // 313 * 16
#define NT1   313    // N tiles for GEMM1 (V)
#define KS1   20     // K steps for GEMM1 (640/32)
#define NT2   160    // N tiles for GEMM2 (2560/16)
#define KS2   40     // K steps for GEMM2 (1280/32)
#define FRAG  512    // elements per fragment: 32 lanes * 16 halves

// ---- WMMA fragment index helpers (from CDNA5 ISA 7.12.2 layout tables) ----
// A matrix 16x32 bf16: lane l (M=l%16), element e -> K offset within 32:
__device__ __forceinline__ int kA(int e, int l) {
  return e + ((e >= 8) ? 8 : 0) + ((l >= 16) ? 8 : 0);
}
// B matrix 32x16 bf16: lane l (N=l%16), element e -> K offset within 32:
__device__ __forceinline__ int kB(int e, int l) {
  return e + ((e >= 8) ? 8 : 0) + ((l >= 16) ? 8 : 0);
}

__device__ __forceinline__ v8f wmma_bf16(v16bf a, v16bf b, v8f c) {
  // D = A(16x32) * B(32x16) + C, f32 accumulate
  return __builtin_amdgcn_wmma_f32_16x16x32_bf16(
      /*neg_a=*/false, a, /*neg_b=*/false, b,
      /*c_mod=*/(short)0, c, /*reuse_a=*/false, /*reuse_b=*/false);
}

__device__ __forceinline__ float fsigmoid(float x) {
  return 1.0f / (1.0f + __expf(-x));
}

// ---------------------------------------------------------------------------
// One-time weight packing kernels
// ---------------------------------------------------------------------------
__global__ void pack_wc_kernel(const float* __restrict__ Wc,
                               __bf16* __restrict__ out) {
  long long idx = (long long)blockIdx.x * blockDim.x + threadIdx.x;
  const long long total = (long long)NT1 * KS1 * FRAG;
  if (idx >= total) return;
  int f   = (int)(idx >> 9);      // fragment index
  int rem = (int)(idx & 511);
  int l = rem >> 4, e = rem & 15;
  int nt = f / KS1, ks = f % KS1;
  int k = ks * 32 + kB(e, l);     // k < 640 always
  int n = nt * 16 + (l & 15);
  float v = (n < VDIM) ? Wc[(long long)k * VDIM + n] : 0.0f;
  out[idx] = (__bf16)v;
}

__global__ void pack_w2_kernel(const float* __restrict__ Wx,
                               const float* __restrict__ Wh,
                               __bf16* __restrict__ out) {
  long long idx = (long long)blockIdx.x * blockDim.x + threadIdx.x;
  const long long total = (long long)NT2 * KS2 * FRAG;
  if (idx >= total) return;
  int f   = (int)(idx >> 9);
  int rem = (int)(idx & 511);
  int l = rem >> 4, e = rem & 15;
  int nt = f / KS2, ks = f % KS2;
  int k = ks * 32 + kB(e, l);     // 0..1279
  int n = nt * 16 + (l & 15);     // 0..2559
  float v = (k < HDIM) ? Wx[(long long)k * H4 + n]
                       : Wh[(long long)(k - HDIM) * H4 + n];
  out[idx] = (__bf16)v;
}

__global__ void pack_embed_kernel(const float* __restrict__ embed,
                                  __bf16* __restrict__ out) {
  long long idx = (long long)blockIdx.x * blockDim.x + threadIdx.x;
  const long long total = (long long)VDIM * HDIM;
  if (idx >= total) return;
  out[idx] = (__bf16)embed[idx];
}

// ---------------------------------------------------------------------------
// Init: h=c=0, tok=BLANK, upd=1, scores=0; pack xcat = [embed[0]; 0]
// grid = 2 blocks x 512 threads; block b owns rows b*16 .. b*16+15
// ---------------------------------------------------------------------------
__global__ void init_kernel(float* __restrict__ h, float* __restrict__ c,
                            float* __restrict__ scores, int* __restrict__ tok,
                            int* __restrict__ upd,
                            const __bf16* __restrict__ embedBf,
                            __bf16* __restrict__ xcat) {
  int b = blockIdx.x;
  if (threadIdx.x < 16) {
    int m = b * 16 + threadIdx.x;
    scores[m] = 0.0f;
    tok[m] = 0;
    upd[m] = 1;
  }
  for (int idx = threadIdx.x; idx < 16 * HDIM; idx += blockDim.x) {
    int m = b * 16 + idx / HDIM;
    int k = idx % HDIM;
    h[m * HDIM + k] = 0.0f;
    c[m * HDIM + k] = 0.0f;
  }
  __syncthreads();
  for (int idx = threadIdx.x; idx < KS2 * FRAG; idx += blockDim.x) {
    int ks = idx >> 9, rem = idx & 511;
    int l = rem >> 4, e = rem & 15;
    int k = ks * 32 + kA(e, l);
    __bf16 v = (k < HDIM) ? embedBf[k] : (__bf16)0.0f;  // tok=0, h=0
    xcat[((b * KS2 + ks) * 32 + l) * 16 + e] = v;
  }
}

// ---------------------------------------------------------------------------
// Generic WMMA bf16 GEMM: C[32 x nTiles*16] = A[32 x kSteps*32] * B + bias
// One 16x16 output tile per wave32. A and B are in packed fragment layout.
// ---------------------------------------------------------------------------
__global__ __launch_bounds__(256)
void gemm_bf16_kernel(const __bf16* __restrict__ Apack,
                      const __bf16* __restrict__ Bpack,
                      const float* __restrict__ bias,
                      float* __restrict__ out,
                      int ldOut, int nTiles, int kSteps, int nBias) {
  int wave = (blockIdx.x * blockDim.x + threadIdx.x) >> 5;
  int lane = threadIdx.x & 31;
  if (wave >= 2 * nTiles) return;
  int nt = wave >> 1;
  int mt = wave & 1;

  const v16bf* a = reinterpret_cast<const v16bf*>(Apack) + (mt * kSteps * 32 + lane);
  const v16bf* b = reinterpret_cast<const v16bf*>(Bpack) + (nt * kSteps * 32 + lane);

  v8f acc = {};
#pragma unroll 4
  for (int ks = 0; ks < kSteps; ++ks) {
    v16bf av = a[ks * 32];
    v16bf bv = b[ks * 32];
    acc = wmma_bf16(av, bv, acc);
  }

  int n = nt * 16 + (lane & 15);
  float bval = (n < nBias) ? bias[n] : 0.0f;
  int mbase = mt * 16 + ((lane >> 4) << 3);   // C/D layout: lanes>=16 -> M+8
#pragma unroll
  for (int r = 0; r < 8; ++r) {
    out[(mbase + r) * ldOut + n] = acc[r] + bval;
  }
}

// ---------------------------------------------------------------------------
// REDUCE: per-row online log-softmax + argmax + token select; then pack
// xcat = [embed[new_tok]; h_old] A-fragments. grid = 2 x 512, warp per row.
// ---------------------------------------------------------------------------
__global__ __launch_bounds__(512)
void reduce_pack_kernel(const float* __restrict__ logits,
                        float* __restrict__ scores, int* __restrict__ tok,
                        int* __restrict__ upd,
                        const __bf16* __restrict__ embedBf,
                        const float* __restrict__ h,
                        __bf16* __restrict__ xcat,
                        float* __restrict__ preds, int t) {
  __shared__ int s_tok[16];
  int b = blockIdx.x;
  int warp = threadIdx.x >> 5;
  int lane = threadIdx.x & 31;
  int m = b * 16 + warp;

  // online softmax pass over logits row
  float mx = -3.4e38f, sm = 0.0f;
  int ag = 0x7fffffff;
  const float* row = logits + (long long)m * VPAD;
  for (int n = lane; n < VDIM; n += 32) {
    float x = row[n];
    if (x > mx) {
      sm = sm * __expf(mx - x) + 1.0f;
      mx = x;
      ag = n;
    } else {
      sm += __expf(x - mx);
    }
  }
  // wave32 combine (keeps first-occurrence argmax on ties)
#pragma unroll
  for (int off = 16; off > 0; off >>= 1) {
    float om = __shfl_xor(mx, off, 32);
    float os = __shfl_xor(sm, off, 32);
    int   oa = __shfl_xor(ag, off, 32);
    if (om > mx || (om == mx && oa < ag)) {
      sm = os + sm * __expf(mx - om);
      mx = om;
      ag = oa;
    } else {
      sm = sm + os * __expf(om - mx);
    }
  }
  if (lane == 0) {
    float val = -__logf(sm);           // max log-softmax = -log(sum exp(x-max))
    int pos = ag;
    int u = (pos != 0);                // BLANK_ID == 0
    if (u) scores[m] += val;
    int ntok = u ? pos : tok[m];
    tok[m] = ntok;
    upd[m] = u;
    preds[(long long)m * TLEN + t] = u ? (float)pos : 0.0f;
    s_tok[warp] = ntok;
  }
  __syncthreads();

  // pack LSTM input [embed[tok]; h] into A fragments for this M-tile
  for (int idx = threadIdx.x; idx < KS2 * FRAG; idx += blockDim.x) {
    int ks = idx >> 9, rem = idx & 511;
    int l = rem >> 4, e = rem & 15;
    int mloc = l & 15;
    int mg = b * 16 + mloc;
    int k = ks * 32 + kA(e, l);
    __bf16 v;
    if (k < HDIM)
      v = embedBf[(long long)s_tok[mloc] * HDIM + k];
    else
      v = (__bf16)h[mg * HDIM + (k - HDIM)];
    xcat[((b * KS2 + ks) * 32 + l) * 16 + e] = v;
  }
}

// ---------------------------------------------------------------------------
// GATES: masked LSTM cell update from gates[32x2560]; then pack
// joint = tanh(tn[:, t_next, :] + out_PN) A-fragments. grid = 2 x 512.
// ---------------------------------------------------------------------------
__global__ __launch_bounds__(512)
void gates_pack_kernel(const float* __restrict__ gates,
                       float* __restrict__ h, float* __restrict__ c,
                       float* __restrict__ outPN, const int* __restrict__ upd,
                       const float* __restrict__ tn,
                       __bf16* __restrict__ jointPack, int t_next) {
  int b = blockIdx.x;
  for (int idx = threadIdx.x; idx < 16 * HDIM; idx += blockDim.x) {
    int mloc = idx / HDIM;
    int k = idx % HDIM;
    int m = b * 16 + mloc;
    if (upd[m]) {
      const float* g = gates + (long long)m * H4;
      float gi = g[k];
      float gf = g[HDIM + k];
      float gg = g[2 * HDIM + k];
      float go = g[3 * HDIM + k];
      float c2 = fsigmoid(gf) * c[m * HDIM + k] + fsigmoid(gi) * tanhf(gg);
      float h2 = fsigmoid(go) * tanhf(c2);
      c[m * HDIM + k] = c2;
      h[m * HDIM + k] = h2;
      outPN[m * HDIM + k] = h2;
    }
  }
  __syncthreads();
  if (t_next >= TLEN) return;
  for (int idx = threadIdx.x; idx < KS1 * FRAG; idx += blockDim.x) {
    int ks = idx >> 9, rem = idx & 511;
    int l = rem >> 4, e = rem & 15;
    int mloc = l & 15;
    int m = b * 16 + mloc;
    int k = ks * 32 + kA(e, l);
    float v = tanhf(tn[((long long)m * TLEN + t_next) * HDIM + k] +
                    outPN[m * HDIM + k]);
    jointPack[((b * KS1 + ks) * 32 + l) * 16 + e] = (__bf16)v;
  }
}

__global__ void copy_scores_kernel(const float* __restrict__ scores,
                                   float* __restrict__ out) {
  if (threadIdx.x < BATCH) out[threadIdx.x] = scores[threadIdx.x];
}

// ---------------------------------------------------------------------------
// Workspace layout (bytes); total ~20.7 MB
// ---------------------------------------------------------------------------
static const size_t WC_OFF  = 0;                                   // 6,410,240
static const size_t W2_OFF  = WC_OFF + (size_t)NT1 * KS1 * FRAG * 2;
static const size_t EMB_OFF = W2_OFF + (size_t)NT2 * KS2 * FRAG * 2;
static const size_t JNT_OFF = EMB_OFF + (size_t)VDIM * HDIM * 2;
static const size_t XC_OFF  = JNT_OFF + (size_t)2 * KS1 * FRAG * 2;
static const size_t LG_OFF  = XC_OFF + (size_t)2 * KS2 * FRAG * 2;
static const size_t GT_OFF  = LG_OFF + (size_t)BATCH * VPAD * 4;
static const size_t H_OFF   = GT_OFF + (size_t)BATCH * H4 * 4;
static const size_t C_OFF   = H_OFF + (size_t)BATCH * HDIM * 4;
static const size_t PN_OFF  = C_OFF + (size_t)BATCH * HDIM * 4;
static const size_t SC_OFF  = PN_OFF + (size_t)BATCH * HDIM * 4;
static const size_t TK_OFF  = SC_OFF + 256;
static const size_t UP_OFF  = TK_OFF + 256;

extern "C" void kernel_launch(void* const* d_in, const int* in_sizes, int n_in,
                              void* d_out, int out_size, void* d_ws, size_t ws_size,
                              hipStream_t stream) {
  (void)in_sizes; (void)n_in; (void)out_size; (void)ws_size;

  const float* tn    = (const float*)d_in[0];  // [B,T,H]
  const float* embed = (const float*)d_in[1];  // [V,H]
  const float* Wx    = (const float*)d_in[2];  // [H,4H]
  const float* Wh    = (const float*)d_in[3];  // [H,4H]
  const float* bb    = (const float*)d_in[4];  // [4H]
  const float* Wc    = (const float*)d_in[5];  // [H,V]
  const float* bc    = (const float*)d_in[6];  // [V]

  char* ws = (char*)d_ws;
  __bf16* wcPack   = (__bf16*)(ws + WC_OFF);
  __bf16* w2Pack   = (__bf16*)(ws + W2_OFF);
  __bf16* embedBf  = (__bf16*)(ws + EMB_OFF);
  __bf16* jointPk  = (__bf16*)(ws + JNT_OFF);
  __bf16* xcatPk   = (__bf16*)(ws + XC_OFF);
  float*  logits   = (float*)(ws + LG_OFF);
  float*  gates    = (float*)(ws + GT_OFF);
  float*  hbuf     = (float*)(ws + H_OFF);
  float*  cbuf     = (float*)(ws + C_OFF);
  float*  pnbuf    = (float*)(ws + PN_OFF);
  float*  scores   = (float*)(ws + SC_OFF);
  int*    tok      = (int*)(ws + TK_OFF);
  int*    updf     = (int*)(ws + UP_OFF);

  float* out_scores = (float*)d_out;       // [32]
  float* out_preds  = out_scores + BATCH;  // [32,1000]

  // --- one-time weight packing (bf16 WMMA fragment layout) ---
  {
    long long n1 = (long long)NT1 * KS1 * FRAG;
    pack_wc_kernel<<<(unsigned)((n1 + 255) / 256), 256, 0, stream>>>(Wc, wcPack);
    long long n2 = (long long)NT2 * KS2 * FRAG;
    pack_w2_kernel<<<(unsigned)((n2 + 255) / 256), 256, 0, stream>>>(Wx, Wh, w2Pack);
    long long n3 = (long long)VDIM * HDIM;
    pack_embed_kernel<<<(unsigned)((n3 + 255) / 256), 256, 0, stream>>>(embed, embedBf);
  }

  // --- state init + initial LSTM step (tok=BLANK, h=c=0, upd=all-true) ---
  init_kernel<<<2, 512, 0, stream>>>(hbuf, cbuf, scores, tok, updf, embedBf, xcatPk);

  const int g1blocks = (2 * NT1 + 7) / 8;  // 8 waves per 256-thread block
  const int g2blocks = (2 * NT2 + 7) / 8;

  // initial PN step: gates = xcat @ W2 + b ; update h,c,out_PN ; pack joint_0
  gemm_bf16_kernel<<<g2blocks, 256, 0, stream>>>(xcatPk, w2Pack, bb, gates,
                                                 H4, NT2, KS2, H4);
  gates_pack_kernel<<<2, 512, 0, stream>>>(gates, hbuf, cbuf, pnbuf, updf, tn,
                                           jointPk, 0);

  // --- main serial decode loop (captured into the hip graph) ---
  for (int t = 0; t < TLEN; ++t) {
    gemm_bf16_kernel<<<g1blocks, 256, 0, stream>>>(jointPk, wcPack, bc, logits,
                                                   VPAD, NT1, KS1, VDIM);
    reduce_pack_kernel<<<2, 512, 0, stream>>>(logits, scores, tok, updf,
                                              embedBf, hbuf, xcatPk,
                                              out_preds, t);
    if (t + 1 < TLEN) {
      gemm_bf16_kernel<<<g2blocks, 256, 0, stream>>>(xcatPk, w2Pack, bb, gates,
                                                     H4, NT2, KS2, H4);
      gates_pack_kernel<<<2, 512, 0, stream>>>(gates, hbuf, cbuf, pnbuf, updf,
                                               tn, jointPk, t + 1);
    }
  }

  copy_scores_kernel<<<1, 32, 0, stream>>>(scores, out_scores);
}
